// ChebyKANLayer_82471962018527
// MI455X (gfx1250) — matrix operand order repaired and checked by reading, not compiled, not run on hardware
//
#include <hip/hip_runtime.h>
#include <hip/hip_bf16.h>

// ---------------------------------------------------------------------------
// ChebyKAN layer as a pipelined bf16 WMMA GEMM on gfx1250 (MI455X).
//   out[n,o] = clip( sum_{i,d} T_d(tanh(x[n,i])) * coeffs[o,i,d] )
// GEMM: A[N, K=5*I] (bf16 features built on the fly via v_tanh_f32)
//       x B[K, O]  (coeffs[O][I][5] flattened == column-major B, k = 5*i+d)
// Roofline: 687 GFLOP vs ~336 MB HBM -> strongly compute bound; all MACs on
// v_wmma_f32_16x16x32_bf16. B tiles staged with GLOBAL_LOAD_ASYNC_TO_LDS_B128
// (ASYNCcnt), A tiles built in registers and double-buffered in LDS.
// BN=256 so the tanh/Chebyshev VALU work (recomputed per column pass) is done
// only O/BN = 8 times. waves_per_eu(2) pins VGPR allocation so 2 waves/SIMD
// always co-resident (one computes while the other stages).
// Note: t = tanh(x) in (-1,1) => all T_d(t) in [-1,1], so the reference's
// +-10 clips inside the recurrence never bind; only the final clip applies.
// ---------------------------------------------------------------------------

typedef __attribute__((ext_vector_type(16))) __bf16 bf16x16;
typedef __attribute__((ext_vector_type(8)))  __bf16 bf16x8;
typedef __attribute__((ext_vector_type(4)))  __bf16 bf16x4;
typedef __attribute__((ext_vector_type(8)))  float  f32x8;

#define N_ROWS 16384
#define N_IN   2048
#define N_OUT  2048
#define KDIM   (N_IN * 5)       // 10240
#define BM     128
#define BN     256
#define BK     160              // 32 x-elements * 5 degrees per K chunk
#define NITER  (KDIM / BK)      // 64
#define LDSS   168              // padded LDS stride (16B aligned, bank-spread)
#define CLAMPV 10.0f

__device__ __forceinline__ float clipf(float v) {
    return fminf(fmaxf(v, -CLAMPV), CLAMPV);
}

// gfx1250 native tanh (TRANS pipe, co-executes with WMMA)
__device__ __forceinline__ float htanh(float x) {
    float r;
    asm("v_tanh_f32 %0, %1" : "=v"(r) : "v"(x));
    return r;
}

// ---------------- Phase 1: coeffs f32 -> bf16 workspace --------------------
__global__ __launch_bounds__(256) void cheby_convert_coeffs(
        const float* __restrict__ src, __bf16* __restrict__ dst, int n) {
    int idx = (blockIdx.x * 256 + threadIdx.x) * 4;
    if (idx + 3 < n) {
        float4 v = *(const float4*)(src + idx);
        bf16x4 o;
        o.x = (__bf16)v.x;
        o.y = (__bf16)v.y;
        o.z = (__bf16)v.z;
        o.w = (__bf16)v.w;
        *(bf16x4*)(dst + idx) = o;
    }
}

// ---- stage helpers --------------------------------------------------------

// Issue one B tile (BN x BK bf16 = 5120 x 16B segments / 256 thr = 20 each)
// as async global->LDS copies. Tracked on ASYNCcnt, no VGPR data.
__device__ __forceinline__ void issue_B_async(
        __bf16 (*BsBuf)[LDSS], const __bf16* __restrict__ Bcol, int kb, int tid) {
    #pragma unroll
    for (int q = 0; q < 20; ++q) {
        int seg  = tid + q * 256;
        int brow = seg / 20;
        int boff = (seg % 20) * 8;
        unsigned lds = (unsigned)(unsigned long long)(void*)&BsBuf[brow][boff];
        const __bf16* g = Bcol + (size_t)brow * KDIM + kb + boff;
        asm volatile("global_load_async_to_lds_b128 %0, %1, off"
                     :: "v"(lds), "v"(g) : "memory");
    }
}

// Build half a row of A features (16 x-elements -> 80 bf16) and store to LDS.
__device__ __forceinline__ void feat_store(const float4 xr[4], __bf16* dst) {
    union { __bf16 e[80]; bf16x8 v[10]; } f;
    #pragma unroll
    for (int q = 0; q < 4; ++q) {
        float xs[4] = {xr[q].x, xr[q].y, xr[q].z, xr[q].w};
        #pragma unroll
        for (int j = 0; j < 4; ++j) {
            float t  = htanh(xs[j]);               // T1
            float T2 = 2.0f * t * t - 1.0f;
            float T3 = 2.0f * t * T2 - t;
            float T4 = 2.0f * t * T3 - T2;
            int b = (q * 4 + j) * 5;
            f.e[b + 0] = (__bf16)1.0f;
            f.e[b + 1] = (__bf16)t;
            f.e[b + 2] = (__bf16)T2;
            f.e[b + 3] = (__bf16)T3;
            f.e[b + 4] = (__bf16)T4;
        }
    }
    #pragma unroll
    for (int q = 0; q < 10; ++q)
        *(bf16x8*)(dst + q * 8) = f.v[q];
}

// ---------------- Phase 2: pipelined feature-build + WMMA GEMM -------------
__global__ __launch_bounds__(256)
__attribute__((amdgpu_waves_per_eu(2)))
void cheby_wmma_gemm(
        const float* __restrict__ X,          // [N_ROWS][N_IN] f32
        const __bf16* __restrict__ Bw,        // [N_OUT][KDIM] bf16
        float* __restrict__ Out) {            // [N_ROWS][N_OUT] f32

    __shared__ __bf16 As[2][BM][LDSS];        // 2 x 43008 B
    __shared__ __bf16 Bs[2][BN][LDSS];        // 2 x 86016 B   (total ~258 KB)

    const int tid  = threadIdx.x;
    const int lane = tid & 31;
    const int wave = tid >> 5;                // 8 waves
    const int wm   = wave & 1;                // 0..1 : 64-row strip
    const int wn   = wave >> 1;               // 0..3 : 64-col strip
    const int m16  = lane & 15;
    const int half = lane >> 4;

    const int gRow0 = blockIdx.y * BM;
    const int gCol0 = blockIdx.x * BN;

    // A staging: 2 threads per row, 16 consecutive x-elements each
    const int arow = tid >> 1;
    const int aoff = (tid & 1) * 16;          // x-element offset within chunk
    const float*  xrow = X  + (size_t)(gRow0 + arow) * N_IN + aoff;
    const __bf16* Bcol = Bw + (size_t)gCol0 * KDIM;

    f32x8  acc[4][4] = {};
    float4 xr[4];

    // ---- prologue: stage iteration 0 into buffer 0 ----
    #pragma unroll
    for (int q = 0; q < 4; ++q)
        xr[q] = *(const float4*)(xrow + q * 4);
    issue_B_async(Bs[0], Bcol, 0, tid);
    feat_store(xr, &As[0][arow][aoff * 5]);
    asm volatile("s_wait_asynccnt 0x0" ::: "memory");
    __syncthreads();

    for (int it = 0; it < NITER; ++it) {
        const int cur = it & 1;
        const int nxt = cur ^ 1;

        // ---- prefetch it+1: x into regs, B async into LDS[nxt] ----
        if (it + 1 < NITER) {
            #pragma unroll
            for (int q = 0; q < 4; ++q)
                xr[q] = *(const float4*)(xrow + (it + 1) * 32 + q * 4);
            issue_B_async(Bs[nxt], Bcol, (it + 1) * BK, tid);
        }

        // ---- compute: 5 WMMA k-steps of 32, 16 tiles per wave (64x64) ----
        #pragma unroll
        for (int c = 0; c < 5; ++c) {
            const int kc = c * 32;
            union { bf16x16 v; bf16x8 h[2]; } af[4], bf[4];
            #pragma unroll
            for (int r = 0; r < 4; ++r) {
                int row = wm * 64 + r * 16 + m16;
                af[r].h[0] = *(const bf16x8*)&As[cur][row][kc + half * 8];
                af[r].h[1] = *(const bf16x8*)&As[cur][row][kc + 16 + half * 8];
            }
            #pragma unroll
            for (int nn = 0; nn < 4; ++nn) {
                int col = wn * 64 + nn * 16 + m16;
                bf[nn].h[0] = *(const bf16x8*)&Bs[cur][col][kc + half * 8];
                bf[nn].h[1] = *(const bf16x8*)&Bs[cur][col][kc + 16 + half * 8];
            }
            #pragma unroll
            for (int r = 0; r < 4; ++r)
                #pragma unroll
                for (int nn = 0; nn < 4; ++nn)
                    acc[r][nn] = __builtin_amdgcn_wmma_f32_16x16x32_bf16(
                        false, af[r].v, false, bf[nn].v,
                        (short)0, acc[r][nn], false, false);
        }

        // ---- tail: build A[nxt] features, ensure B[nxt] landed ----
        if (it + 1 < NITER) {
            feat_store(xr, &As[nxt][arow][aoff * 5]);
            asm volatile("s_wait_asynccnt 0x0" ::: "memory");
        }
        __syncthreads();
    }

    // ---- epilogue: clamp + store (C/D layout: M = half*8 + j, N = m16) ----
    #pragma unroll
    for (int r = 0; r < 4; ++r) {
        #pragma unroll
        for (int nn = 0; nn < 4; ++nn) {
            int col = gCol0 + wn * 64 + nn * 16 + m16;
            #pragma unroll
            for (int j = 0; j < 8; ++j) {
                int row = gRow0 + wm * 64 + r * 16 + half * 8 + j;
                Out[(size_t)row * N_OUT + col] = clipf(acc[r][nn][j]);
            }
        }
    }
}

// ---------------------------------------------------------------------------
extern "C" void kernel_launch(void* const* d_in, const int* in_sizes, int n_in,
                              void* d_out, int out_size, void* d_ws, size_t ws_size,
                              hipStream_t stream) {
    const float* x      = (const float*)d_in[0];   // [16384, 2048] f32
    const float* coeffs = (const float*)d_in[1];   // [2048, 2048, 5] f32
    float* out          = (float*)d_out;           // [16384, 2048] f32
    __bf16* Bw          = (__bf16*)d_ws;           // 40 MB bf16 B matrix

    // Phase 1: coeffs -> bf16 (layout already column-major B: [O][K])
    const int nCoef = N_OUT * KDIM;                // 20,971,520
    cheby_convert_coeffs<<<nCoef / (256 * 4), 256, 0, stream>>>(coeffs, Bw, nCoef);

    // Phase 2: pipelined feature-build + WMMA GEMM
    dim3 grid(N_OUT / BN, N_ROWS / BM);            // (8, 128)
    cheby_wmma_gemm<<<grid, 256, 0, stream>>>(x, Bw, out);
}
